// MambaPPO_18098992185578
// MI455X (gfx1250) — matrix operand order repaired and checked by reading, not compiled, not run on hardware
//
#include <hip/hip_runtime.h>
#include <hip/hip_bf16.h>
#include <stdint.h>

// ---------------------------------------------------------------------------
// MI455X / gfx1250: wave32, WMMA f32_16x16x32_f16 for all contractions,
// Tensor Data Mover (tensor_load_to_lds) staging for the conv GEMM weights,
// LDS-staged on-the-fly A_bar generation (exp2 deduplicated across waves).
// ---------------------------------------------------------------------------

typedef _Float16 half8 __attribute__((ext_vector_type(8)));
typedef _Float16 v16h  __attribute__((ext_vector_type(16)));
typedef float    v8f   __attribute__((ext_vector_type(8)));
typedef unsigned int u32x4 __attribute__((ext_vector_type(4)));
typedef int      i32x4 __attribute__((ext_vector_type(4)));
typedef int      i32x8 __attribute__((ext_vector_type(8)));

#define B_    64
#define N_    256
#define HID_  512
#define INNER 1024
#define STATE 1024
#define TOK   (B_ * N_)        // 16384 tokens
#define TWOIN (2 * INNER)      // 2048

// Conv weight panel in LDS: 64 rows x 128 halves, padded to 68 dwords (272 B)
// per row so ds_load_b128 stays 16B-aligned and fragment rows avoid conflicts.
#define PANEL_K   128
#define ROW_DW    68
#define PANEL_DW  (64 * ROW_DW)

// A_bar panel in LDS: 128 cols x 32 halves, pitch 20 dwords (80 B) per col.
#define AB_PITCH_DW 20
#define AB_PANEL_DW (128 * AB_PITCH_DW)

// ---------------- WMMA fragment helpers (contiguous k = 16*h + s, same in A and B) ----------------

__device__ __forceinline__ v16h ld_a_row(const _Float16* rowp, int lane) {
  int hi = lane >> 4;
  const _Float16* p = rowp + hi * 16;
  half8 lo = *(const half8*)(p);
  half8 hh = *(const half8*)(p + 8);
  v16h r;
#pragma unroll
  for (int i = 0; i < 8; ++i) { r[i] = lo[i]; r[i + 8] = hh[i]; }
  return r;
}

// B (32x16) fragment from N-major (transposed) weights BT[n][k].
__device__ __forceinline__ v16h ld_bT(const _Float16* BT, int ldb, int lane) {
  int n = lane & 15, hi = lane >> 4;
  const _Float16* p = BT + (size_t)n * ldb + hi * 16;
  half8 lo = *(const half8*)(p);
  half8 hh = *(const half8*)(p + 8);
  v16h r;
#pragma unroll
  for (int i = 0; i < 8; ++i) { r[i] = lo[i]; r[i + 8] = hh[i]; }
  return r;
}

__device__ __forceinline__ v16h ld_b_2x16B(const _Float16* p0,
                                           const _Float16* p1) {
  half8 lo = *(const half8*)p0;
  half8 hh = *(const half8*)p1;
  v16h r;
#pragma unroll
  for (int i = 0; i < 8; ++i) { r[i] = lo[i]; r[i + 8] = hh[i]; }
  return r;
}

__device__ __forceinline__ v8f wmma16(v16h a, v16h b, v8f c) {
  return __builtin_amdgcn_wmma_f32_16x16x32_f16(false, a, false, b,
                                                (short)0, c, false, false);
}

__device__ __forceinline__ float silu_f(float v) {
  return v / (1.0f + __expf(-v));
}

// ---------------- stage 0: LayerNorm (one wave per token) -------------------

__global__ __launch_bounds__(256) void mamba_ln_kernel(
    const float* __restrict__ x, const float* __restrict__ gamma,
    const float* __restrict__ beta, _Float16* __restrict__ h16) {
  int lane = threadIdx.x & 31, wave = threadIdx.x >> 5;
  int token = blockIdx.x * 8 + wave;
  const float* row = x + (size_t)token * HID_;
  float v[16];
  float s = 0.f, ss = 0.f;
#pragma unroll
  for (int j = 0; j < 16; ++j) {
    float t = row[lane + 32 * j];
    v[j] = t; s += t; ss += t * t;
  }
#pragma unroll
  for (int off = 16; off > 0; off >>= 1) {
    s  += __shfl_xor(s,  off, 32);
    ss += __shfl_xor(ss, off, 32);
  }
  float mu  = s * (1.0f / HID_);
  float var = ss * (1.0f / HID_) - mu * mu;
  float rstd = rsqrtf(var + 1e-3f);
  _Float16* orow = h16 + (size_t)token * HID_;
#pragma unroll
  for (int j = 0; j < 16; ++j) {
    int idx = lane + 32 * j;
    orow[idx] = (_Float16)((v[j] - mu) * rstd * gamma[idx] + beta[idx]);
  }
}

// ---------------- weight prep: transpose + convert --------------------------

__global__ void mamba_tcvt_kernel(const float* __restrict__ in,
                                  _Float16* __restrict__ out, int R, int C) {
  int idx = blockIdx.x * blockDim.x + threadIdx.x;
  if (idx >= R * C) return;
  int r = idx / C, c = idx % C;
  out[(size_t)c * R + r] = (_Float16)in[idx];
}

__global__ void mamba_softplusT_kernel(const float* __restrict__ A,
                                       float* __restrict__ ST, int R, int C) {
  int idx = blockIdx.x * blockDim.x + threadIdx.x;
  if (idx >= R * C) return;
  int r = idx / C, c = idx % C;
  float a = A[idx];
  float sp = fmaxf(a, 0.0f) + log1pf(__expf(-fabsf(a)));
  ST[(size_t)c * R + r] = sp;
}

// ---------------- GEMM1: h16 @ W_in^T + b_in -> u|gate (f16) ----------------
// Unroll-by-2 ping-pong: no cross-set register copies, loads for one k-step
// stay in flight under the other k-step's WMMAs.

__global__ __launch_bounds__(256) void mamba_gemm_in_kernel(
    const _Float16* __restrict__ h16, const _Float16* __restrict__ WinT,
    const float* __restrict__ b_in, _Float16* __restrict__ h2) {
  int lane = threadIdx.x & 31, wave = threadIdx.x >> 5;
  int mt = blockIdx.x * 8 + wave;
  int n0 = blockIdx.y * 64;
  const _Float16* Arow = h16 + ((size_t)mt * 16 + (lane & 15)) * HID_;
  v8f acc[4] = {};
  v16h aA = ld_a_row(Arow, lane);
  v16h bA[4];
#pragma unroll
  for (int j = 0; j < 4; ++j)
    bA[j] = ld_bT(WinT + (size_t)(n0 + 16 * j) * HID_, HID_, lane);
  for (int k0 = 0; k0 < HID_; k0 += 64) {
    v16h aB = ld_a_row(Arow + k0 + 32, lane);
    v16h bB[4];
#pragma unroll
    for (int j = 0; j < 4; ++j)
      bB[j] = ld_bT(WinT + (size_t)(n0 + 16 * j) * HID_ + k0 + 32, HID_, lane);
#pragma unroll
    for (int j = 0; j < 4; ++j) acc[j] = wmma16(aA, bA[j], acc[j]);
    if (k0 + 64 < HID_) {
      aA = ld_a_row(Arow + k0 + 64, lane);
#pragma unroll
      for (int j = 0; j < 4; ++j)
        bA[j] = ld_bT(WinT + (size_t)(n0 + 16 * j) * HID_ + k0 + 64, HID_, lane);
    }
#pragma unroll
    for (int j = 0; j < 4; ++j) acc[j] = wmma16(aB, bB[j], acc[j]);
  }
  int hi = lane >> 4, nn = lane & 15;
#pragma unroll
  for (int j = 0; j < 4; ++j) {
    int col = n0 + 16 * j + nn;
    float bias = b_in[col];
#pragma unroll
    for (int i = 0; i < 8; ++i) {
      int m = i + hi * 8;
      h2[((size_t)(mt * 16 + m)) * TWOIN + col] = (_Float16)(acc[j][i] + bias);
    }
  }
}

// ---------------- GEMM2: causal conv = 4 shifted GEMMs, silu ----------------
// Weights staged through the Tensor Data Mover into LDS, double buffered.

__global__ __launch_bounds__(256) void mamba_gemm_conv_kernel(
    const _Float16* __restrict__ h2 /* u = cols 0..INNER-1 */,
    const _Float16* __restrict__ WcT, _Float16* __restrict__ uc) {
  __shared__ __attribute__((aligned(16))) uint32_t smem[2 * PANEL_DW];
  int lane = threadIdx.x & 31, wave = threadIdx.x >> 5;
  int mt = blockIdx.x * 8 + wave;
  int n0 = blockIdx.y * 64;
  int m = lane & 15, hi = lane >> 4;
  int t = mt * 16 + m;                  // this lane's token (A row)
  int npos = t & (N_ - 1);

  v8f acc[4] = {};

  // panel p: tap = p>>3, k-offset = (p&7)*128; D# per CDNA5 ISA §8.3/8.4
  auto issue_tdm = [&](int p) {
    int tap = p >> 3, kp = (p & 7) * PANEL_K;
    const _Float16* gsrc =
        WcT + (size_t)tap * INNER * INNER + (size_t)n0 * INNER + kp;
    uint64_t ga = (uint64_t)(uintptr_t)gsrc;
    uint32_t lds = (uint32_t)(uintptr_t)&smem[(p & 1) * PANEL_DW];
    u32x4 g0;
    i32x8 g1;
    i32x4 g2 = {}, g3 = {};
    g0[0] = 1u;                                        // count=1, user D#
    g0[1] = lds;                                       // lds_addr
    g0[2] = (uint32_t)ga;                              // global_addr lo
    g0[3] = (uint32_t)((ga >> 32) & 0x01FFFFFFu) | 0x80000000u; // hi | type=2
    // data_size=2B, pad_enable, pad_interval=64dw, pad_amount=4dw
    g1[0] = (1 << 16) | (1 << 20) | (5 << 22) | (3 << 25);
    g1[1] = (int)((1024u & 0xFFFFu) << 16);            // tensor_dim0 lo
    g1[2] = (int)(1024u << 16);                        // dim0 hi | dim1 lo
    g1[3] = (int)((uint32_t)PANEL_K << 16);            // dim1 hi | tile_dim0
    g1[4] = 64;                                        // tile_dim1 | tile_dim2
    g1[5] = 1024;                                      // tensor_dim0_stride lo
    g1[6] = 0;
    g1[7] = 0;
#if defined(__clang_major__) && __clang_major__ >= 23
    i32x8 g4 = {};
    __builtin_amdgcn_tensor_load_to_lds(g0, g1, g2, g3, g4, 0);
#else
    __builtin_amdgcn_tensor_load_to_lds(g0, g1, g2, g3, 0);
#endif
  };

  if (wave == 0) issue_tdm(0);

  const int P = 32;                     // 4 taps x 8 k-panels
  for (int p = 0; p < P; ++p) {
    int tap = p >> 3;
    if (wave == 0) {
      if (p + 1 < P) {
        issue_tdm(p + 1);
        __builtin_amdgcn_s_wait_tensorcnt(1);   // panel p landed (in-order)
      } else {
        __builtin_amdgcn_s_wait_tensorcnt(0);
      }
    }
    __syncthreads();

    bool valid = (npos + tap - 3) >= 0;
    const _Float16* Arow =
        h2 + (long long)(t + tap - 3) * TWOIN + (p & 7) * PANEL_K;
    const uint32_t* sb = &smem[(p & 1) * PANEL_DW];
#pragma unroll
    for (int ks = 0; ks < 4; ++ks) {
      v16h a = {};
      if (valid) a = ld_a_row(Arow + ks * 32, lane);
#pragma unroll
      for (int j = 0; j < 4; ++j) {
        const _Float16* pb =
            (const _Float16*)(sb + (size_t)(16 * j + (lane & 15)) * ROW_DW) +
            ks * 32 + hi * 16;
        v16h b = ld_b_2x16B(pb, pb + 8);
        acc[j] = wmma16(a, b, acc[j]);
      }
    }
    __syncthreads();
  }

  int nn = lane & 15;
#pragma unroll
  for (int j = 0; j < 4; ++j) {
    int col = n0 + 16 * j + nn;
#pragma unroll
    for (int i = 0; i < 8; ++i) {
      int mm = i + hi * 8;
      uc[((size_t)(mt * 16 + mm)) * INNER + col] = (_Float16)silu_f(acc[j][i]);
    }
  }
}

// ------- GEMM3: y[b,n,h] = sum_d uc[b,n,d] * exp(S[d,h]*(n+1)), * silu(gate)
// A_bar panels (128 cols x 32 k, f16) are generated cooperatively into LDS
// once per block (exp2 deduplicated 4x), double-buffered: S-loads for panel
// k+1 are issued before the WMMAs on panel k.

__global__ __launch_bounds__(256) void mamba_gemm_abar_kernel(
    const _Float16* __restrict__ uc, const float* __restrict__ ST,
    const _Float16* __restrict__ h2 /* gate = cols INNER.. */,
    _Float16* __restrict__ y) {
  __shared__ __attribute__((aligned(16))) uint32_t ab[2 * AB_PANEL_DW];
  int tid = threadIdx.x;
  int lane = tid & 31, wave = tid >> 5;
  int n = blockIdx.y;                   // sequence position 0..255
  int mt = wave & 3;                    // batch tile (4 x 16 = 64)
  int cg = wave >> 2;                   // column group 0..1
  int h0blk = blockIdx.x * 128;
  int h0 = h0blk + cg * 64;
  int m = lane & 15, hi = lane >> 4, nn = lane & 15;
  int b = mt * 16 + m;
  const _Float16* Arow = uc + ((size_t)(b * N_ + n)) * INNER;
  float c1 = (float)(n + 1) * 1.4426950408889634f;  // pos * log2(e)

  // generator role: two threads per column; 16 exps each per panel
  int gcol = tid >> 1;
  int goff = (tid & 1) * 16;
  const float* gsp = ST + (size_t)(h0blk + gcol) * INNER + goff;

  v8f acc[4] = {};

  {  // panel 0
    _Float16* d = (_Float16*)&ab[0] + gcol * (AB_PITCH_DW * 2) + goff;
#pragma unroll
    for (int e = 0; e < 16; ++e) d[e] = (_Float16)exp2f(gsp[e] * c1);
  }
  __syncthreads();

  for (int ks = 0; ks < 32; ++ks) {
    int k0 = ks * 32;
    int cur = ks & 1, nxt = cur ^ 1;
    bool have = (ks + 1) < 32;
    float sv[16];
    if (have) {
      const float* sp = gsp + (ks + 1) * 32;
#pragma unroll
      for (int e = 0; e < 16; ++e) sv[e] = sp[e];
    }
    v16h a = ld_a_row(Arow + k0, lane);
    const uint32_t* bbuf = &ab[cur * AB_PANEL_DW];
#pragma unroll
    for (int j = 0; j < 4; ++j) {
      int col = cg * 64 + 16 * j + nn;
      const _Float16* pb =
          (const _Float16*)(bbuf + (size_t)col * AB_PITCH_DW) + hi * 16;
      v16h bf = ld_b_2x16B(pb, pb + 8);
      acc[j] = wmma16(a, bf, acc[j]);
    }
    if (have) {
      _Float16* d =
          (_Float16*)&ab[nxt * AB_PANEL_DW] + gcol * (AB_PITCH_DW * 2) + goff;
#pragma unroll
      for (int e = 0; e < 16; ++e) d[e] = (_Float16)exp2f(sv[e] * c1);
    }
    __syncthreads();
  }

#pragma unroll
  for (int j = 0; j < 4; ++j) {
    int hcol = h0 + 16 * j + nn;
#pragma unroll
    for (int i = 0; i < 8; ++i) {
      int bb = mt * 16 + i + hi * 8;
      size_t tok = (size_t)bb * N_ + n;
      float g = (float)h2[tok * TWOIN + INNER + hcol];
      y[tok * INNER + hcol] = (_Float16)(acc[j][i] * silu_f(g));
    }
  }
}

// ---------------- GEMM4: y @ W_out^T + b_out + skip -> fp32 out -------------

__global__ __launch_bounds__(256) void mamba_gemm_out_kernel(
    const _Float16* __restrict__ y, const _Float16* __restrict__ WoutT,
    const float* __restrict__ b_out, const float* __restrict__ x,
    float* __restrict__ out) {
  int lane = threadIdx.x & 31, wave = threadIdx.x >> 5;
  int mt = blockIdx.x * 8 + wave;
  int n0 = blockIdx.y * 64;
  const _Float16* Arow = y + ((size_t)mt * 16 + (lane & 15)) * STATE;
  v8f acc[4] = {};
  v16h aA = ld_a_row(Arow, lane);
  v16h bA[4];
#pragma unroll
  for (int j = 0; j < 4; ++j)
    bA[j] = ld_bT(WoutT + (size_t)(n0 + 16 * j) * STATE, STATE, lane);
  for (int k0 = 0; k0 < STATE; k0 += 64) {
    v16h aB = ld_a_row(Arow + k0 + 32, lane);
    v16h bB[4];
#pragma unroll
    for (int j = 0; j < 4; ++j)
      bB[j] = ld_bT(WoutT + (size_t)(n0 + 16 * j) * STATE + k0 + 32, STATE, lane);
#pragma unroll
    for (int j = 0; j < 4; ++j) acc[j] = wmma16(aA, bA[j], acc[j]);
    if (k0 + 64 < STATE) {
      aA = ld_a_row(Arow + k0 + 64, lane);
#pragma unroll
      for (int j = 0; j < 4; ++j)
        bA[j] = ld_bT(WoutT + (size_t)(n0 + 16 * j) * STATE + k0 + 64, STATE, lane);
    }
#pragma unroll
    for (int j = 0; j < 4; ++j) acc[j] = wmma16(aB, bB[j], acc[j]);
  }
  int hi = lane >> 4, nn = lane & 15;
#pragma unroll
  for (int j = 0; j < 4; ++j) {
    int col = n0 + 16 * j + nn;
    float bias = b_out[col];
#pragma unroll
    for (int i = 0; i < 8; ++i) {
      int mm = i + hi * 8;
      size_t tok = (size_t)(mt * 16 + mm);
      out[tok * HID_ + col] = acc[j][i] + bias + x[tok * HID_ + col];
    }
  }
}

// ---------------- workspace layout ------------------------------------------

static constexpr size_t OFF_H16   = 0;                                   // 16 MB
static constexpr size_t OFF_H2    = OFF_H16  + (size_t)TOK * HID_  * 2;  // 64 MB
static constexpr size_t OFF_UC    = OFF_H2   + (size_t)TOK * TWOIN * 2;  // 32 MB
static constexpr size_t OFF_Y     = OFF_UC   + (size_t)TOK * INNER * 2;  // 32 MB
static constexpr size_t OFF_WINT  = OFF_Y    + (size_t)TOK * INNER * 2;  //  2 MB
static constexpr size_t OFF_WCT   = OFF_WINT + (size_t)TWOIN * HID_ * 2; //  8 MB
static constexpr size_t OFF_WOUTT = OFF_WCT  + (size_t)4 * INNER * INNER * 2;
static constexpr size_t OFF_ST    = OFF_WOUTT+ (size_t)HID_ * STATE * 2; //  4 MB

extern "C" void kernel_launch(void* const* d_in, const int* in_sizes, int n_in,
                              void* d_out, int out_size, void* d_ws, size_t ws_size,
                              hipStream_t stream) {
  const float* x      = (const float*)d_in[0];
  const float* gamma  = (const float*)d_in[1];
  const float* beta   = (const float*)d_in[2];
  const float* W_in   = (const float*)d_in[3];
  const float* b_in   = (const float*)d_in[4];
  const float* W_conv = (const float*)d_in[5];
  const float* A      = (const float*)d_in[6];
  const float* W_out  = (const float*)d_in[7];
  const float* b_out  = (const float*)d_in[8];
  float* out = (float*)d_out;

  uint8_t* ws = (uint8_t*)d_ws;
  _Float16* h16   = (_Float16*)(ws + OFF_H16);
  _Float16* h2    = (_Float16*)(ws + OFF_H2);
  _Float16* uc    = (_Float16*)(ws + OFF_UC);
  _Float16* ybuf  = (_Float16*)(ws + OFF_Y);
  _Float16* WinT  = (_Float16*)(ws + OFF_WINT);
  _Float16* WcT   = (_Float16*)(ws + OFF_WCT);
  _Float16* WoutT = (_Float16*)(ws + OFF_WOUTT);
  float*    ST    = (float*)   (ws + OFF_ST);

  // stage 0: layernorm -> f16
  mamba_ln_kernel<<<dim3(TOK / 8), 256, 0, stream>>>(x, gamma, beta, h16);

  // weight prep (recomputed every call: deterministic, graph-capture safe)
  mamba_tcvt_kernel<<<(HID_ * TWOIN + 255) / 256, 256, 0, stream>>>(
      W_in, WinT, HID_, TWOIN);
  for (int k = 0; k < 4; ++k)
    mamba_tcvt_kernel<<<(INNER * INNER + 255) / 256, 256, 0, stream>>>(
        W_conv + (size_t)k * INNER * INNER, WcT + (size_t)k * INNER * INNER,
        INNER, INNER);
  mamba_tcvt_kernel<<<(STATE * HID_ + 255) / 256, 256, 0, stream>>>(
      W_out, WoutT, STATE, HID_);
  mamba_softplusT_kernel<<<(INNER * STATE + 255) / 256, 256, 0, stream>>>(
      A, ST, INNER, STATE);

  // GEMM pipeline
  mamba_gemm_in_kernel<<<dim3(TOK / 128, TWOIN / 64), 256, 0, stream>>>(
      h16, WinT, b_in, h2);
  mamba_gemm_conv_kernel<<<dim3(TOK / 128, INNER / 64), 256, 0, stream>>>(
      h2, WcT, uc);
  mamba_gemm_abar_kernel<<<dim3(STATE / 128, N_), 256, 0, stream>>>(
      uc, ST, h2, ybuf);
  mamba_gemm_out_kernel<<<dim3(TOK / 128, HID_ / 64), 256, 0, stream>>>(
      ybuf, WoutT, b_out, x, out);
}
